// GCNRR_44461501448669
// MI455X (gfx1250) — matrix operand order repaired and checked by reading, not compiled, not run on hardware
//
#include <hip/hip_runtime.h>
#include <hip/hip_bf16.h>

// ---------------------------------------------------------------------------
// Two-layer GCN forward for MI455X (gfx1250, wave32).
//   h1  = x @ W1                      (bf16 WMMA, f32 accum)
//   a1  = segsum(norm * h1[src]) + b1 (self-loop folded in, f32 atomics in L2)
//   h2  = a1 @ W2                     (bf16 WMMA, f32 accum)
//   out = segsum(norm * h2[src]) + b2
// ---------------------------------------------------------------------------

typedef __attribute__((ext_vector_type(16))) __bf16 v16bf;
typedef __attribute__((ext_vector_type(8)))  __bf16 v8bf;
typedef __attribute__((ext_vector_type(8)))  float  v8f;

// ---------------- degree / normalization ----------------

__global__ void gcn_init_deg(float* deg, int n) {
    int i = blockIdx.x * blockDim.x + threadIdx.x;
    if (i < n) deg[i] = 1.0f;   // self-loop contribution
}

__global__ void gcn_add_deg(const int* __restrict__ dst, float* deg, long long e) {
    long long i = (long long)blockIdx.x * blockDim.x + threadIdx.x;
    if (i < e) atomicAdd(&deg[dst[i]], 1.0f);
}

__global__ void gcn_deg_to_dis(float* deg, int n) {
    int i = blockIdx.x * blockDim.x + threadIdx.x;
    if (i < n) deg[i] = rsqrtf(deg[i]);   // deg >= 1 always (self-loop)
}

// ---------------- conversions ----------------

__global__ void gcn_f32_to_bf16(const float* __restrict__ in, __bf16* __restrict__ out, long long n) {
    long long i = (long long)blockIdx.x * blockDim.x + threadIdx.x;
    if (i < n) out[i] = (__bf16)in[i];
}

// W [K,N] row-major f32  ->  Wt [N,K] (column-major of W) bf16
__global__ void gcn_w_transpose_bf16(const float* __restrict__ W, __bf16* __restrict__ Wt, int K, int N) {
    int i = blockIdx.x * blockDim.x + threadIdx.x;
    if (i >= K * N) return;
    int k = i / N, n = i % N;
    Wt[(long long)n * K + k] = (__bf16)W[i];
}

// agg_bias_bf16: out = bf16(in + bias[c])
__global__ void gcn_bias_to_bf16(const float* __restrict__ in, const float* __restrict__ bias,
                                 __bf16* __restrict__ out, int n, int C) {
    long long i = (long long)blockIdx.x * blockDim.x + threadIdx.x;
    if (i >= (long long)n * C) return;
    out[i] = (__bf16)(in[i] + bias[(int)(i % C)]);
}

// self-loop init: agg[i,c] = dis[i]^2 * h[i,c] (+ bias[c])
__global__ void gcn_self_init(const float* __restrict__ h, const float* __restrict__ dis,
                              const float* __restrict__ bias, float* __restrict__ agg,
                              int n, int C, int use_bias) {
    long long i = (long long)blockIdx.x * blockDim.x + threadIdx.x;
    if (i >= (long long)n * C) return;
    int node = (int)(i / C);
    int c    = (int)(i % C);
    float d = dis[node];
    float v = d * d * h[i];
    if (use_bias) v += bias[c];
    agg[i] = v;
}

// ---------------- edge scatter (one wave32 per edge) ----------------

__global__ void gcn_scatter_edges(const float* __restrict__ h,
                                  const int* __restrict__ src, const int* __restrict__ dst,
                                  const float* __restrict__ dis, float* __restrict__ agg,
                                  int C, long long E) {
    long long w = ((long long)blockIdx.x * blockDim.x + threadIdx.x) >> 5;
    if (w >= E) return;
    int lane = threadIdx.x & 31;
    int s = src[w];
    int d = dst[w];
    float nrm = dis[s] * dis[d];
    const float* hs = h + (long long)s * C;
    float*       ad = agg + (long long)d * C;
    // h, agg working sets (<= 51 MB each) are L2-resident on MI455X (192 MB L2):
    // these float atomics never leave the chip.
    for (int c = lane; c < C; c += 32)
        atomicAdd(&ad[c], nrm * hs[c]);
}

// ---------------- WMMA GEMM ----------------
// C[M,ND] = A[M,KD] (bf16 row-major) x Bt (bf16, [ND,KD] i.e. B column-major)
// One wave per 16-row tile; wave computes the full 16 x ND strip.
//
// Fragment layouts used by this kernel (wave32, contiguous k order for A and B):
//  A 16x32 bf16 : lanes 0-15 row M=lane,   elems 0-15 -> K k0+0..15
//                 lanes 16-31 row M=lane-16, elems 0-15 -> K k0+16..31
//  B 32x16 bf16 : lanes 0-15 col N=lane,   elems 0-15 -> K k0+0..15
//                 lanes 16-31 col N=lane-16, elems 0-15 -> K k0+16..31
//  D 16x16 f32  : VGPR v: lanes 0-15 -> (row v, col lane); lanes 16-31 -> (row v+8, col lane-16)
template <int KD, int ND>
__global__ void gcn_wmma_gemm(const __bf16* __restrict__ A,
                              const __bf16* __restrict__ Bt,
                              float* __restrict__ C, int mtiles) {
    int wave = (int)(((long long)blockIdx.x * blockDim.x + threadIdx.x) >> 5);
    if (wave >= mtiles) return;
    int lane = threadIdx.x & 31;
    int half = lane >> 4;     // 0: lanes 0-15, 1: lanes 16-31
    int l15  = lane & 15;

    constexpr int NT = ND / 16;
    v8f acc[NT];
#pragma unroll
    for (int t = 0; t < NT; ++t)
        acc[t] = (v8f){0.f, 0.f, 0.f, 0.f, 0.f, 0.f, 0.f, 0.f};

    const __bf16* arow = A + (long long)(wave * 16 + l15) * KD;

#pragma unroll
    for (int k0 = 0; k0 < KD; k0 += 32) {
        // ---- A fragment ----
        v8bf alo = *(const v8bf*)(arow + k0 + half * 16);       // K k0 + half*16 + 0..7
        v8bf ahi = *(const v8bf*)(arow + k0 + half * 16 + 8);   // K k0 + half*16 + 8..15
        v16bf af;
#pragma unroll
        for (int i = 0; i < 8; ++i) { af[i] = alo[i]; af[i + 8] = ahi[i]; }

#pragma unroll
        for (int t = 0; t < NT; ++t) {
            // ---- B fragment (from transposed weights: contiguous K run per lane) ----
            const __bf16* bcol = Bt + (long long)(t * 16 + l15) * KD + k0 + half * 16;
            v8bf blo = *(const v8bf*)(bcol);
            v8bf bhi = *(const v8bf*)(bcol + 8);
            v16bf bf;
#pragma unroll
            for (int i = 0; i < 8; ++i) { bf[i] = blo[i]; bf[i + 8] = bhi[i]; }

            acc[t] = __builtin_amdgcn_wmma_f32_16x16x32_bf16(
                false, af, false, bf, (short)0, acc[t], false, false);
        }
    }

    // ---- store D ----
#pragma unroll
    for (int t = 0; t < NT; ++t) {
        float* crow = C + (long long)(wave * 16 + half * 8) * ND + t * 16 + l15;
#pragma unroll
        for (int v = 0; v < 8; ++v)
            crow[(long long)v * ND] = acc[t][v];
    }
}

// ---------------------------------------------------------------------------

extern "C" void kernel_launch(void* const* d_in, const int* in_sizes, int n_in,
                              void* d_out, int out_size, void* d_ws, size_t ws_size,
                              hipStream_t stream) {
    (void)n_in; (void)out_size; (void)ws_size;

    const float* x  = (const float*)d_in[0];
    const int*   ei = (const int*)d_in[1];   // edge_index [2,E] (jax default int32)
    const float* W1 = (const float*)d_in[2];
    const float* b1 = (const float*)d_in[3];
    const float* W2 = (const float*)d_in[4];
    const float* b2 = (const float*)d_in[5];
    float* out = (float*)d_out;

    const int IN_C = 256, HID_C = 128, OUT_C = 64;
    const int N = in_sizes[0] / IN_C;          // 100000 (divisible by 16)
    const long long E = in_sizes[1] / 2;       // 1600000
    const int* srcI = ei;
    const int* dstI = ei + E;

    // ---- workspace carve-out (~103 MB, regions reused across phases) ----
    char* ws = (char*)d_ws;
    size_t off = 0;
    auto take = [&](size_t bytes) {
        size_t o = off;
        off += (bytes + 255) & ~(size_t)255;
        return o;
    };
    float*  dis  = (float*)(ws + take((size_t)N * 4));
    __bf16* W1t  = (__bf16*)(ws + take((size_t)IN_C * HID_C * 2));
    __bf16* W2t  = (__bf16*)(ws + take((size_t)HID_C * OUT_C * 2));
    // regionA: x_bf16 (N*256*2) then reused for agg1 (N*128*4) -- same byte size
    char* regionA = ws + take((size_t)N * IN_C * 2);
    // regionB: h1 (N*128*4) then reused for h2in_bf16 (N*128*2) + h2 (N*64*4)
    char* regionB = ws + take((size_t)N * HID_C * 4);

    __bf16* x_bf = (__bf16*)regionA;
    float*  agg1 = (float*)regionA;
    float*  h1   = (float*)regionB;
    __bf16* h2in = (__bf16*)regionB;
    float*  h2   = (float*)(regionB + (size_t)N * HID_C * 2);

    const int B = 256;
    const int mtiles = N / 16;                       // 6250
    const int gemmBlocks = (mtiles + 7) / 8;         // 8 waves / block
    const int edgeBlocks = (int)((E * 32 + B - 1) / B);

    // 1) degree -> dis = deg^-1/2
    gcn_init_deg<<<(N + B - 1) / B, B, 0, stream>>>(dis, N);
    gcn_add_deg<<<(int)((E + B - 1) / B), B, 0, stream>>>(dstI, dis, E);
    gcn_deg_to_dis<<<(N + B - 1) / B, B, 0, stream>>>(dis, N);

    // 2) weight transpose+convert, x convert
    gcn_w_transpose_bf16<<<(IN_C * HID_C + B - 1) / B, B, 0, stream>>>(W1, W1t, IN_C, HID_C);
    gcn_w_transpose_bf16<<<(HID_C * OUT_C + B - 1) / B, B, 0, stream>>>(W2, W2t, HID_C, OUT_C);
    gcn_f32_to_bf16<<<(int)(((long long)N * IN_C + B - 1) / B), B, 0, stream>>>(x, x_bf, (long long)N * IN_C);

    // 3) layer 1: GEMM -> self-loop init -> edge scatter -> +bias, to bf16
    gcn_wmma_gemm<256, 128><<<gemmBlocks, B, 0, stream>>>(x_bf, W1t, h1, mtiles);
    gcn_self_init<<<(int)(((long long)N * HID_C + B - 1) / B), B, 0, stream>>>(
        h1, dis, nullptr, agg1, N, HID_C, 0);
    gcn_scatter_edges<<<edgeBlocks, B, 0, stream>>>(h1, srcI, dstI, dis, agg1, HID_C, E);
    gcn_bias_to_bf16<<<(int)(((long long)N * HID_C + B - 1) / B), B, 0, stream>>>(
        agg1, b1, h2in, N, HID_C);

    // 4) layer 2: GEMM -> self-loop init (+bias) into d_out -> edge scatter into d_out
    gcn_wmma_gemm<128, 64><<<gemmBlocks, B, 0, stream>>>(h2in, W2t, h2, mtiles);
    gcn_self_init<<<(int)(((long long)N * OUT_C + B - 1) / B), B, 0, stream>>>(
        h2, dis, b2, out, N, OUT_C, 1);
    gcn_scatter_edges<<<edgeBlocks, B, 0, stream>>>(h2, srcI, dstI, dis, out, OUT_C, E);
}
